// SpatialGraphConvolution_21251498180686
// MI455X (gfx1250) — compile-verified, matrix operand order checked by reading, not run on hardware
//
#include <hip/hip_runtime.h>
#include <hip/hip_bf16.h>

// ---------------------------------------------------------------------------
// Problem constants (fixed by reference setup_inputs)
// ---------------------------------------------------------------------------
static constexpr int kB  = 4;
static constexpr int kN  = 10000;
static constexpr int kE  = 320000;
static constexpr int kET = kE + kN;      // edges + self loops = 330000
static constexpr int kBN = kB * kN;      // 40000 rows total
static constexpr int kF0 = 256;
static constexpr int kF1 = 256;
static constexpr int kF2 = 128;
static constexpr int kK  = 256;          // GEMM K for both layers
static constexpr float kNegSlope = 0.2f;
static constexpr float kEps = 1e-16f;

typedef __attribute__((ext_vector_type(8)))  float  v8f;
typedef __attribute__((ext_vector_type(8)))  __bf16 v8bf;
typedef __attribute__((ext_vector_type(16))) __bf16 v16bf;

// ---------------------------------------------------------------------------
// Utility kernels
// ---------------------------------------------------------------------------
__global__ void f32_to_bf16_kernel(const float* __restrict__ in,
                                   __bf16* __restrict__ out, int n) {
  int i = blockIdx.x * blockDim.x + threadIdx.x;
  if (i < n) out[i] = (__bf16)in[i];
}

__global__ void fill_i32_kernel(int* __restrict__ p, int v, int n) {
  int i = blockIdx.x * blockDim.x + threadIdx.x;
  if (i < n) p[i] = v;
}

__global__ void copy_i32_kernel(const int* __restrict__ in,
                                int* __restrict__ out, int n) {
  int i = blockIdx.x * blockDim.x + threadIdx.x;
  if (i < n) out[i] = in[i];
}

__global__ void build_edges_kernel(const long long* __restrict__ ei,
                                   int* __restrict__ src,
                                   int* __restrict__ dst, int e, int n) {
  int i = blockIdx.x * blockDim.x + threadIdx.x;
  if (i < e) {
    src[i] = (int)ei[i];
    dst[i] = (int)ei[e + i];
  } else if (i < e + n) {
    int v = i - e;       // self loops appended, as GATConv does
    src[i] = v;
    dst[i] = v;
  }
}

// ---------------------------------------------------------------------------
// CSR build: in-degree histogram -> exclusive scan -> slot fill
// ---------------------------------------------------------------------------
__global__ void degree_kernel(const int* __restrict__ dst,
                              int* __restrict__ cnt, int et) {
  int i = blockIdx.x * blockDim.x + threadIdx.x;
  if (i < et) atomicAdd(&cnt[dst[i]], 1);
}

__global__ void scan_kernel(const int* __restrict__ cnt,
                            int* __restrict__ offs, int n) {
  __shared__ int part[1024];
  __shared__ int stotal;
  const int t  = threadIdx.x;
  const int ck = (n + 1023) >> 10;
  const int beg = t * ck;
  const int end = (beg + ck < n) ? (beg + ck) : n;
  int s = 0;
  for (int i = beg; i < end; ++i) s += cnt[i];
  part[t] = s;
  __syncthreads();
  if (t == 0) {
    int run = 0;
    for (int i = 0; i < 1024; ++i) {
      int v = part[i];
      part[i] = run;
      run += v;
    }
    stotal = run;
  }
  __syncthreads();
  int run = part[t];
  for (int i = beg; i < end; ++i) {
    offs[i] = run;
    run += cnt[i];
  }
  if (t == 0) offs[n] = stotal;
}

__global__ void csr_fill_kernel(const int* __restrict__ dst,
                                int* __restrict__ cursor,
                                int* __restrict__ eids, int et) {
  int i = blockIdx.x * blockDim.x + threadIdx.x;
  if (i >= et) return;
  int pos = atomicAdd(&cursor[dst[i]], 1);
  eids[pos] = i;
}

// ---------------------------------------------------------------------------
// WMMA GEMM:  C[M,Nout] = A[M,K](bf16) * W[Nout,K](bf16)^T   (f32 accumulate)
// One wave32 per 16x16 tile via v_wmma_f32_16x16x32_bf16. The 16x256 weight
// tile is shared by the 4 waves of a block: staged once into LDS with
// global_load_async_to_lds_b128 (ASYNCcnt-tracked), padded to avoid bank
// conflicts on the per-lane ds_load_b128 fragment reads.
// A-fragment lane layout (16-bit A 16x32, ISA 7.12.2):
//   lane l (l<16): row=l,   K = {0..7, 16..23}
//   lane l (>=16): row=l-16,K = {8..15, 24..31}
// ---------------------------------------------------------------------------
static constexpr int kLdsRow = kK + 8;   // 264 bf16 -> 528 B row stride

__global__ void gemm_bf16_wmma_kernel(const __bf16* __restrict__ A,
                                      const __bf16* __restrict__ W,
                                      float* __restrict__ C,
                                      int M, int Nout) {
  __shared__ __align__(16) __bf16 wtile[16 * kLdsRow];

  const int ntile = blockIdx.y;
  const int tid   = threadIdx.y * 32 + threadIdx.x;   // 0..127

  // ---- async stage of W[ntile*16 .. +16, 0..K) into LDS -----------------
  {
    const char* gbase = (const char*)(W + (size_t)ntile * 16 * kK);
    unsigned    lbase = (unsigned)(uintptr_t)(&wtile[0]);
    // 16 rows x 512B = 512 chunks of 16B; 128 threads -> 4 iterations
#pragma unroll
    for (int it = 0; it < 4; ++it) {
      int chunk = it * 128 + tid;
      int row = chunk >> 5;          // 0..15
      int col = chunk & 31;          // 16B chunk within row
      unsigned lds_addr =
          lbase + (unsigned)(row * (kLdsRow * 2) + col * 16);
      unsigned long long gaddr =
          (unsigned long long)(uintptr_t)(gbase + row * 512 + col * 16);
      asm volatile("global_load_async_to_lds_b128 %0, %1, off"
                   :: "v"(lds_addr), "v"(gaddr)
                   : "memory");
    }
    asm volatile("s_wait_asynccnt 0" ::: "memory");
    __syncthreads();
  }

  const int mtile = blockIdx.x * blockDim.y + threadIdx.y;
  if (mtile * 16 >= M) return;

  const int lane  = threadIdx.x & 31;
  const int half  = lane >> 4;    // 0: K 0..7 / 16..23, 1: K 8..15 / 24..31
  const int idx16 = lane & 15;    // row (for A) / column (for B)

  const __bf16* arow = A + (size_t)(mtile * 16 + idx16) * kK + half * 8;
  const __bf16* brow = wtile + idx16 * kLdsRow + half * 8;

  v8f acc = {};
#pragma unroll
  for (int k = 0; k < kK; k += 32) {
    v8bf alo = *(const v8bf*)(arow + k);
    v8bf ahi = *(const v8bf*)(arow + k + 16);
    v8bf blo = *(const v8bf*)(brow + k);
    v8bf bhi = *(const v8bf*)(brow + k + 16);
    v16bf af = __builtin_shufflevector(alo, ahi, 0, 1, 2, 3, 4, 5, 6, 7,
                                       8, 9, 10, 11, 12, 13, 14, 15);
    v16bf bf = __builtin_shufflevector(blo, bhi, 0, 1, 2, 3, 4, 5, 6, 7,
                                       8, 9, 10, 11, 12, 13, 14, 15);
    acc = __builtin_amdgcn_wmma_f32_16x16x32_bf16(
        /*neg_a=*/false, af, /*neg_b=*/false, bf,
        /*c_mod=*/(short)0, acc, /*reuse_a=*/false, /*reuse_b=*/false);
  }

  // C/D layout: VGPR r -> row (r + 8*half), column idx16.
  float* crow =
      C + (size_t)(mtile * 16 + half * 8) * Nout + ntile * 16 + idx16;
#pragma unroll
  for (int r = 0; r < 8; ++r) crow[(size_t)r * Nout] = acc[r];
}

// ---------------------------------------------------------------------------
// Per-row attention logits: als[r] = h[r,:].a1 ; ald[r] = h[r,:].a2
// ---------------------------------------------------------------------------
__global__ void row_dots_kernel(const float* __restrict__ h,
                                const float* __restrict__ a1,
                                const float* __restrict__ a2,
                                float* __restrict__ als,
                                float* __restrict__ ald, int rows, int F) {
  int row = blockIdx.x * blockDim.y + threadIdx.y;
  if (row >= rows) return;
  int lane = threadIdx.x;
  const float* hr = h + (size_t)row * F;
  float s = 0.f, d = 0.f;
  for (int f = lane; f < F; f += 32) {
    float v = hr[f];
    s += v * a1[f];
    d += v * a2[f];
  }
#pragma unroll
  for (int off = 16; off > 0; off >>= 1) {
    s += __shfl_xor(s, off, 32);
    d += __shfl_xor(d, off, 32);
  }
  if (lane == 0) {
    als[row] = s;
    ald[row] = d;
  }
}

__device__ inline float leaky(float e) {
  return (e > 0.f) ? e : kNegSlope * e;
}

// ---------------------------------------------------------------------------
// Fused per-destination softmax + gather-aggregate + bias + ReLU.
// One wave32 per (batch, node). CSR (offs/eids) lists incoming edges.
//   pass 1: m      = max_j  leaky(als[src_j] + ald[node])      (lanes over j)
//   pass 2: denom  = sum_j  exp(e_j - m)                        (lanes over j)
//   pass 3: out[:] = sum_j  alpha_j * h[src_j,:] (+bias, relu)  (lanes over F)
// No atomics; gather loads are 128B coalesced per edge row.
// ---------------------------------------------------------------------------
template <int F, bool BF16OUT>
__global__ void node_gather_kernel(const int* __restrict__ offs,
                                   const int* __restrict__ eids,
                                   const int* __restrict__ srcI,
                                   const float* __restrict__ als,
                                   const float* __restrict__ ald,
                                   const float* __restrict__ h,
                                   const float* __restrict__ bias,
                                   void* __restrict__ outv, int n) {
  int widx = blockIdx.x * blockDim.y + threadIdx.y;   // over B*N
  if (widx >= kB * n) return;
  const int b    = widx / n;
  const int node = widx - b * n;
  const int lane = threadIdx.x;

  const int beg = offs[node];
  const int end = offs[node + 1];
  const float aldn = ald[(size_t)b * n + node];
  const float* alsb = als + (size_t)b * n;

  // pass 1: segment max
  float m = -3.402823466e+38f;
  for (int j = beg + lane; j < end; j += 32) {
    int s = srcI[eids[j]];
    m = fmaxf(m, leaky(alsb[s] + aldn));
  }
#pragma unroll
  for (int off = 16; off > 0; off >>= 1) m = fmaxf(m, __shfl_xor(m, off, 32));

  // pass 2: denominator
  float dsum = 0.f;
  for (int j = beg + lane; j < end; j += 32) {
    int s = srcI[eids[j]];
    dsum += __expf(leaky(alsb[s] + aldn) - m);
  }
#pragma unroll
  for (int off = 16; off > 0; off >>= 1) dsum += __shfl_xor(dsum, off, 32);
  const float inv = 1.f / (dsum + kEps);

  // pass 3: weighted gather (lanes across features, edges sequential)
  constexpr int C = F / 32;
  float acc[C];
#pragma unroll
  for (int c = 0; c < C; ++c) acc[c] = 0.f;
  for (int j = beg; j < end; ++j) {
    int s = srcI[eids[j]];
    float alpha = __expf(leaky(alsb[s] + aldn) - m) * inv;
    const float* hs = h + ((size_t)b * n + s) * F;
#pragma unroll
    for (int c = 0; c < C; ++c) acc[c] += alpha * hs[lane + 32 * c];
  }

  // epilogue: bias + relu
  const size_t obase = ((size_t)b * n + node) * F;
#pragma unroll
  for (int c = 0; c < C; ++c) {
    float v = acc[c] + bias[lane + 32 * c];
    v = (v > 0.f) ? v : 0.f;
    if (BF16OUT)
      ((__bf16*)outv)[obase + lane + 32 * c] = (__bf16)v;
    else
      ((float*)outv)[obase + lane + 32 * c] = v;
  }
}

// ---------------------------------------------------------------------------
// Launch
// ---------------------------------------------------------------------------
extern "C" void kernel_launch(void* const* d_in, const int* in_sizes, int n_in,
                              void* d_out, int out_size, void* d_ws,
                              size_t ws_size, hipStream_t stream) {
  (void)in_sizes; (void)n_in; (void)out_size; (void)ws_size;

  const float*     x   = (const float*)d_in[0];
  const long long* ei  = (const long long*)d_in[1];
  const float*     W0  = (const float*)d_in[2];
  const float*     b0  = (const float*)d_in[3];
  const float*     as0 = (const float*)d_in[4];
  const float*     ad0 = (const float*)d_in[5];
  const float*     W1  = (const float*)d_in[6];
  const float*     b1  = (const float*)d_in[7];
  const float*     as1 = (const float*)d_in[8];
  const float*     ad1 = (const float*)d_in[9];
  float*           out = (float*)d_out;

  // Workspace carve (256B aligned)
  char* ws = (char*)d_ws;
  size_t off = 0;
  auto carve = [&](size_t bytes) -> void* {
    void* p = ws + off;
    off += (bytes + 255) & ~(size_t)255;
    return p;
  };
  __bf16* xbf   = (__bf16*)carve((size_t)kBN * kF0 * sizeof(__bf16));
  __bf16* W0bf  = (__bf16*)carve((size_t)kF1 * kF0 * sizeof(__bf16));
  __bf16* W1bf  = (__bf16*)carve((size_t)kF2 * kF1 * sizeof(__bf16));
  __bf16* h1bf  = (__bf16*)carve((size_t)kBN * kF1 * sizeof(__bf16));
  float*  h     = (float*)carve((size_t)kBN * kF1 * sizeof(float));
  float*  als   = (float*)carve((size_t)kBN * sizeof(float));
  float*  ald   = (float*)carve((size_t)kBN * sizeof(float));
  int*    srcI  = (int*)carve((size_t)kET * sizeof(int));
  int*    dstI  = (int*)carve((size_t)kET * sizeof(int));
  int*    cnt   = (int*)carve((size_t)kN * sizeof(int));
  int*    offs  = (int*)carve((size_t)(kN + 1) * sizeof(int));
  int*    curs  = (int*)carve((size_t)kN * sizeof(int));
  int*    eids  = (int*)carve((size_t)kET * sizeof(int));

  const int TPB = 256;

  // --- prep: bf16 conversions, edge list, CSR ---------------------------
  {
    int n = kBN * kF0;
    f32_to_bf16_kernel<<<(n + TPB - 1) / TPB, TPB, 0, stream>>>(x, xbf, n);
    n = kF1 * kF0;
    f32_to_bf16_kernel<<<(n + TPB - 1) / TPB, TPB, 0, stream>>>(W0, W0bf, n);
    n = kF2 * kF1;
    f32_to_bf16_kernel<<<(n + TPB - 1) / TPB, TPB, 0, stream>>>(W1, W1bf, n);

    build_edges_kernel<<<(kET + TPB - 1) / TPB, TPB, 0, stream>>>(ei, srcI,
                                                                  dstI, kE,
                                                                  kN);
    fill_i32_kernel<<<(kN + TPB - 1) / TPB, TPB, 0, stream>>>(cnt, 0, kN);
    degree_kernel<<<(kET + TPB - 1) / TPB, TPB, 0, stream>>>(dstI, cnt, kET);
    scan_kernel<<<1, 1024, 0, stream>>>(cnt, offs, kN);
    copy_i32_kernel<<<(kN + TPB - 1) / TPB, TPB, 0, stream>>>(offs, curs, kN);
    csr_fill_kernel<<<(kET + TPB - 1) / TPB, TPB, 0, stream>>>(dstI, curs,
                                                               eids, kET);
  }

  // ======================= Layer 0 (256 -> 256) =========================
  {
    dim3 blk(32, 4);
    dim3 grd(kBN / 16 / 4, kF1 / 16);  // 625 x 16
    gemm_bf16_wmma_kernel<<<grd, blk, 0, stream>>>(xbf, W0bf, h, kBN, kF1);
  }
  {
    dim3 blk(32, 8);
    row_dots_kernel<<<(kBN + 7) / 8, blk, 0, stream>>>(h, as0, ad0, als, ald,
                                                       kBN, kF1);
  }
  {
    dim3 blk(32, 8);
    node_gather_kernel<kF1, true><<<(kBN + 7) / 8, blk, 0, stream>>>(
        offs, eids, srcI, als, ald, h, b0, (void*)h1bf, kN);
  }

  // ======================= Layer 1 (256 -> 128) =========================
  {
    dim3 blk(32, 4);
    dim3 grd(kBN / 16 / 4, kF2 / 16);  // 625 x 8
    gemm_bf16_wmma_kernel<<<grd, blk, 0, stream>>>(h1bf, W1bf, h, kBN, kF2);
  }
  {
    dim3 blk(32, 8);
    row_dots_kernel<<<(kBN + 7) / 8, blk, 0, stream>>>(h, as1, ad1, als, ald,
                                                       kBN, kF2);
  }
  {
    dim3 blk(32, 8);
    node_gather_kernel<kF2, false><<<(kBN + 7) / 8, blk, 0, stream>>>(
        offs, eids, srcI, als, ald, h, b1, (void*)out, kN);
  }
}
